// EPAN_sel_8735963480601
// MI455X (gfx1250) — matrix operand order repaired and checked by reading, not compiled, not run on hardware
//
#include <hip/hip_runtime.h>
#include <hip/hip_bf16.h>

// ---------------------------------------------------------------------------
// Problem constants (from reference): B=8, N=4096, C=512, H=8, P=64, D=64
// ---------------------------------------------------------------------------
#define BB   8
#define NN   4096
#define CC   512
#define HH   8
#define PP   64
#define DD   64
#define BNT  32768LL          // B*N
#define HNT  32768            // H*N

// Use CDNA5 async global->LDS copies (ASYNCcnt path). Set to 0 to fall back to
// register-staged ds_store path.
#define USE_ASYNC_LDS 1

typedef __attribute__((ext_vector_type(16))) __bf16 v16bf;
typedef __attribute__((ext_vector_type(8)))  float  v8f;

union FragBF { v16bf v; uint4 q[2]; };

// A-fragment (16x32 bf16, MxK): lane<16 holds K {0..7, 16..23}, lane>=16 holds
// K {8..15, 24..31}; element j<8 -> K = 8*half + j ; j>=8 -> K = 16+8*half+(j-8)
__device__ __forceinline__ v16bf frag_a_lds(const __bf16* row, int half) {
  FragBF f;
  f.q[0] = *(const uint4*)(row + 8 * half);
  f.q[1] = *(const uint4*)(row + 16 + 8 * half);
  return f.v;
}
// B-fragment (32x16 bf16, KxN): lane<16 holds K {0..7, 16..23}, lane>=16 holds
// K {8..15, 24..31} (column = lane%16), same K order as the A-fragment.
__device__ __forceinline__ v16bf frag_b_lds(const __bf16* row, int half) {
  FragBF f;
  f.q[0] = *(const uint4*)(row + 8 * half);
  f.q[1] = *(const uint4*)(row + 16 + 8 * half);
  return f.v;
}

#if USE_ASYNC_LDS
// 16B async copy global -> LDS. VDST carries the LDS byte address (low 32 bits
// of the flat shared-aperture pointer), VADDR the 64-bit global address.
// Tracked by ASYNCcnt.
__device__ __forceinline__ void async_copy_b128(const void* lds_dst, const void* g) {
  unsigned int lds_byte = (unsigned int)(unsigned long long)lds_dst;
  asm volatile("global_load_async_to_lds_b128 %0, %1, off"
               :: "v"(lds_byte), "v"(g) : "memory");
}
__device__ __forceinline__ void wait_async0() {
  asm volatile("s_wait_asynccnt 0x0" ::: "memory");
}
#endif

// ---------------------------------------------------------------------------
// fp32 -> bf16 convert (grid-stride)
// ---------------------------------------------------------------------------
__global__ __launch_bounds__(256) void cvt_f32_bf16_kernel(
    const float* __restrict__ s, __bf16* __restrict__ d, long long n) {
  long long i = (long long)blockIdx.x * blockDim.x + threadIdx.x;
  long long stride = (long long)gridDim.x * blockDim.x;
  for (; i < n; i += stride) d[i] = (__bf16)s[i];
}

// ---------------------------------------------------------------------------
// Generic WMMA bf16 GEMM:  C[M,N] = A[M,K] @ B[N,K]^T  (+bias[n])
// Block: 256 threads = 8 waves (2x4), tile 128x128, k-step 32,
// double-buffered LDS fed by async global->LDS copies (one barrier / k-step).
// MODE 0: bf16 C, row-major   MODE 1: bf16 C, transposed store   MODE 2: f32 C
// ---------------------------------------------------------------------------
template <int MODE>
__global__ __launch_bounds__(256) void gemm_bt_kernel(
    const __bf16* __restrict__ A, const __bf16* __restrict__ Bm,
    void* __restrict__ Cout, const float* __restrict__ bias,
    int M, int Nn, int K, int lda, int ldb, int ldc,
    long long sA, long long sB, long long sC) {
  __shared__ __bf16 As[2][128 * 40];   // 32 data + 8 pad, 80B rows (16B aligned)
  __shared__ __bf16 Bs[2][128 * 40];

  const int z = blockIdx.z;
  A  += (long long)z * sA;
  Bm += (long long)z * sB;
  const int m0 = blockIdx.y * 128;
  const int n0 = blockIdx.x * 128;
  const int t = threadIdx.x;
  const int lane = t & 31, wid = t >> 5;
  const int wm = wid & 1, wn = wid >> 1;   // wave grid 2(M) x 4(N)
  const int half = lane >> 4, l16 = lane & 15;

  // fixed per-thread copy slots: 512 uint4 per matrix per k-step, 2 per thread
  const int row0 = t >> 2, seg = t & 3;
  const int row1 = row0 + 64;
  // clamp OOB rows to last valid row: loads stay in-bounds; the garbage only
  // feeds C elements that the store guards discard.
  const int ga0 = (m0 + row0 < M)  ? m0 + row0 : M - 1;
  const int ga1 = (m0 + row1 < M)  ? m0 + row1 : M - 1;
  const int gb0 = (n0 + row0 < Nn) ? n0 + row0 : Nn - 1;
  const int gb1 = (n0 + row1 < Nn) ? n0 + row1 : Nn - 1;
  const __bf16* Ar0 = A  + (long long)ga0 * lda + seg * 8;
  const __bf16* Ar1 = A  + (long long)ga1 * lda + seg * 8;
  const __bf16* Br0 = Bm + (long long)gb0 * ldb + seg * 8;
  const __bf16* Br1 = Bm + (long long)gb1 * ldb + seg * 8;

  v8f acc[4][2];
#pragma unroll
  for (int i = 0; i < 4; ++i)
#pragma unroll
    for (int j = 0; j < 2; ++j)
      acc[i][j] = (v8f){0.f, 0.f, 0.f, 0.f, 0.f, 0.f, 0.f, 0.f};

  const int kSteps = K >> 5;

#if USE_ASYNC_LDS
  {
    // prologue: tile 0 into buffer 0
    async_copy_b128(&As[0][row0 * 40 + seg * 8], Ar0);
    async_copy_b128(&As[0][row1 * 40 + seg * 8], Ar1);
    async_copy_b128(&Bs[0][row0 * 40 + seg * 8], Br0);
    async_copy_b128(&Bs[0][row1 * 40 + seg * 8], Br1);
    wait_async0();
    __syncthreads();
  }
  for (int ks = 0; ks < kSteps; ++ks) {
    const int buf = ks & 1;
    if (ks + 1 < kSteps) {                 // issue next tile into other buffer
      const long long kc = (long long)(ks + 1) * 32;
      async_copy_b128(&As[buf ^ 1][row0 * 40 + seg * 8], Ar0 + kc);
      async_copy_b128(&As[buf ^ 1][row1 * 40 + seg * 8], Ar1 + kc);
      async_copy_b128(&Bs[buf ^ 1][row0 * 40 + seg * 8], Br0 + kc);
      async_copy_b128(&Bs[buf ^ 1][row1 * 40 + seg * 8], Br1 + kc);
    }
    v16bf af[4], bf[2];
#pragma unroll
    for (int mi = 0; mi < 4; ++mi)
      af[mi] = frag_a_lds(&As[buf][(wm * 64 + mi * 16 + l16) * 40], half);
#pragma unroll
    for (int ni = 0; ni < 2; ++ni)
      bf[ni] = frag_b_lds(&Bs[buf][(wn * 32 + ni * 16 + l16) * 40], half);
#pragma unroll
    for (int mi = 0; mi < 4; ++mi)
#pragma unroll
      for (int ni = 0; ni < 2; ++ni)
        acc[mi][ni] = __builtin_amdgcn_wmma_f32_16x16x32_bf16(
            false, af[mi], false, bf[ni], (short)0, acc[mi][ni], false, false);
    wait_async0();      // next tile resident
    __syncthreads();    // all waves done reading current buffer
  }
#else
  for (int ks = 0; ks < kSteps; ++ks) {
    const int buf = 0;
    const long long kc = (long long)ks * 32;
    uint4 a0 = *(const uint4*)(Ar0 + kc), a1 = *(const uint4*)(Ar1 + kc);
    uint4 b0 = *(const uint4*)(Br0 + kc), b1 = *(const uint4*)(Br1 + kc);
    *(uint4*)(&As[buf][row0 * 40 + seg * 8]) = a0;
    *(uint4*)(&As[buf][row1 * 40 + seg * 8]) = a1;
    *(uint4*)(&Bs[buf][row0 * 40 + seg * 8]) = b0;
    *(uint4*)(&Bs[buf][row1 * 40 + seg * 8]) = b1;
    __syncthreads();
    v16bf af[4], bf[2];
#pragma unroll
    for (int mi = 0; mi < 4; ++mi)
      af[mi] = frag_a_lds(&As[buf][(wm * 64 + mi * 16 + l16) * 40], half);
#pragma unroll
    for (int ni = 0; ni < 2; ++ni)
      bf[ni] = frag_b_lds(&Bs[buf][(wn * 32 + ni * 16 + l16) * 40], half);
#pragma unroll
    for (int mi = 0; mi < 4; ++mi)
#pragma unroll
      for (int ni = 0; ni < 2; ++ni)
        acc[mi][ni] = __builtin_amdgcn_wmma_f32_16x16x32_bf16(
            false, af[mi], false, bf[ni], (short)0, acc[mi][ni], false, false);
    __syncthreads();
  }
#endif

  // epilogue: C layout -> lane holds col = lane%16, rows = r + 8*(lane/16)
#pragma unroll
  for (int mi = 0; mi < 4; ++mi) {
#pragma unroll
    for (int ni = 0; ni < 2; ++ni) {
#pragma unroll
      for (int r = 0; r < 8; ++r) {
        int gm = m0 + wm * 64 + mi * 16 + r + 8 * half;
        int gn = n0 + wn * 32 + ni * 16 + l16;
        if (gm < M && gn < Nn) {
          float v = acc[mi][ni][r];
          if (bias) v += bias[gn];
          if (MODE == 0)
            ((__bf16*)Cout)[(long long)z * sC + (long long)gm * ldc + gn] = (__bf16)v;
          else if (MODE == 1)
            ((__bf16*)Cout)[(long long)z * sC + (long long)gn * ldc + gm] = (__bf16)v;
          else
            ((float*)Cout)[(long long)z * sC + (long long)gm * ldc + gn] = v;
        }
      }
    }
  }
}

// ---------------------------------------------------------------------------
// inv-L2-norm of q over tokens: invn[b*512 + h*64 + d] = 1/max(||q||, 1e-12)
// ---------------------------------------------------------------------------
__global__ __launch_bounds__(256) void qnorm_kernel(
    const __bf16* __restrict__ qkv, float* __restrict__ invn) {
  const int idx = blockIdx.x;         // b*512 + c, c = h*64 + d
  const int b = idx >> 9, c = idx & 511;
  __shared__ float red[256];
  float s = 0.f;
  for (int n = threadIdx.x; n < NN; n += 256) {
    float q = (float)qkv[((long long)b * NN + n) * 1536 + c];
    s += q * q;
  }
  red[threadIdx.x] = s;
  __syncthreads();
  for (int o = 128; o; o >>= 1) {
    if (threadIdx.x < o) red[threadIdx.x] += red[threadIdx.x + o];
    __syncthreads();
  }
  if (threadIdx.x == 0)
    invn[idx] = 1.f / fmaxf(sqrtf(red[0]), 1e-12f);
}

// ---------------------------------------------------------------------------
// repack: qn[b,h,n,d] (normalized), kT/vT[b,h,d,n], xT[b,c,n]
// ---------------------------------------------------------------------------
__global__ __launch_bounds__(256) void repack_kernel(
    const __bf16* __restrict__ qkv, const __bf16* __restrict__ xbf,
    const float* __restrict__ invn, __bf16* __restrict__ qn,
    __bf16* __restrict__ kT, __bf16* __restrict__ vT, __bf16* __restrict__ xT) {
  long long i = (long long)blockIdx.x * 256 + threadIdx.x;  // over B*N*C
  if (i >= BNT * CC) return;
  const int c = (int)(i & 511);
  const long long bn = i >> 9;
  const int n = (int)(bn & 4095);
  const int b = (int)(bn >> 12);
  const int h = c >> 6, d = c & 63;
  const long long base = bn * 1536;
  const float q = (float)qkv[base + c];
  const float k = (float)qkv[base + 512 + c];
  const float v = (float)qkv[base + 1024 + c];
  const long long bh = b * HH + h;
  qn[(bh * NN + n) * 64 + d]  = (__bf16)(q * invn[b * 512 + c]);
  kT[(bh * 64 + d) * NN + n]  = (__bf16)k;
  vT[(bh * 64 + d) * NN + n]  = (__bf16)v;
  xT[((long long)b * CC + c) * NN + n] = xbf[i];
}

// ---------------------------------------------------------------------------
// Fused projected spatial attention:
//   scores = (qn @ kp) * temp2[h]; probs = softmax_P(scores); out = probs @ vp^T
//   epilogue applies the faithful scrambled permute (0,3,1,2)+reshape.
// grid: (N/128, B*H), 256 threads = 8 waves, each wave handles 16 rows of N.
// ---------------------------------------------------------------------------
__global__ __launch_bounds__(256) void attn_sa_kernel(
    const __bf16* __restrict__ qn, const __bf16* __restrict__ kpT,
    const __bf16* __restrict__ vp, const float* __restrict__ temp2,
    __bf16* __restrict__ x_sa) {
  __shared__ __bf16 skp[64 * 72];        // [p][d]
  __shared__ __bf16 svp[64 * 72];        // [d][p]
  __shared__ __bf16 sprob[8 * 16 * 72];  // per-wave [16][72] re-layout pad

  const int bh = blockIdx.y;
  const int b = bh >> 3, h = bh & 7;
  const int t = threadIdx.x, lane = t & 31, wid = t >> 5;
  const int half = lane >> 4, l16 = lane & 15;

  const __bf16* kpb = kpT + (long long)bh * 64 * 64;
  const __bf16* vpb = vp  + (long long)bh * 64 * 64;
#pragma unroll
  for (int s = 0; s < 2; ++s) {
    int slot = t + s * 256;
    int row = slot >> 3, seg = slot & 7;
#if USE_ASYNC_LDS
    async_copy_b128(skp + row * 72 + seg * 8, kpb + row * 64 + seg * 8);
    async_copy_b128(svp + row * 72 + seg * 8, vpb + row * 64 + seg * 8);
#else
    *(uint4*)(skp + row * 72 + seg * 8) = *(const uint4*)(kpb + row * 64 + seg * 8);
    *(uint4*)(svp + row * 72 + seg * 8) = *(const uint4*)(vpb + row * 64 + seg * 8);
#endif
  }
#if USE_ASYNC_LDS
  wait_async0();
#endif
  __syncthreads();

  const float scale = temp2[h];
  const int nrow = blockIdx.x * 128 + wid * 16 + l16;  // row this lane feeds A with
  const __bf16* qrow = qn + ((long long)bh * NN + nrow) * 64;

  // A-fragments of qn rows (K = D = 64 -> 2 k-blocks), direct from global
  v16bf aq[2];
#pragma unroll
  for (int kb = 0; kb < 2; ++kb) {
    FragBF f;
    f.q[0] = *(const uint4*)(qrow + kb * 32 + 8 * half);
    f.q[1] = *(const uint4*)(qrow + kb * 32 + 16 + 8 * half);
    aq[kb] = f.v;
  }

  // scores[n, p]: B[k=d][col=p] = kpT[p][d]
  v8f sc[4];
#pragma unroll
  for (int pf = 0; pf < 4; ++pf)
    sc[pf] = (v8f){0.f, 0.f, 0.f, 0.f, 0.f, 0.f, 0.f, 0.f};
#pragma unroll
  for (int pf = 0; pf < 4; ++pf)
#pragma unroll
    for (int kb = 0; kb < 2; ++kb) {
      v16bf bk = frag_b_lds(skp + (pf * 16 + l16) * 72 + kb * 32, half);
      sc[pf] = __builtin_amdgcn_wmma_f32_16x16x32_bf16(
          false, aq[kb], false, bk, (short)0, sc[pf], false, false);
    }

  // softmax over P=64: values for row r live in 16 lanes (same half) x 4 frags
  float sv[4][8], e[4][8], rmax[8], rsum[8];
#pragma unroll
  for (int r = 0; r < 8; ++r) {
    rmax[r] = -3.4e38f;
#pragma unroll
    for (int pf = 0; pf < 4; ++pf) {
      sv[pf][r] = sc[pf][r] * scale;
      rmax[r] = fmaxf(rmax[r], sv[pf][r]);
    }
  }
#pragma unroll
  for (int off = 1; off < 16; off <<= 1)
#pragma unroll
    for (int r = 0; r < 8; ++r)
      rmax[r] = fmaxf(rmax[r], __shfl_xor(rmax[r], off, 32));
#pragma unroll
  for (int r = 0; r < 8; ++r) {
    rsum[r] = 0.f;
#pragma unroll
    for (int pf = 0; pf < 4; ++pf) {
      e[pf][r] = __expf(sv[pf][r] - rmax[r]);
      rsum[r] += e[pf][r];
    }
  }
#pragma unroll
  for (int off = 1; off < 16; off <<= 1)
#pragma unroll
    for (int r = 0; r < 8; ++r)
      rsum[r] += __shfl_xor(rsum[r], off, 32);

  // write probs (bf16) into wave-private LDS pad in row-major [16][72]
  __bf16* pw = sprob + wid * (16 * 72);
#pragma unroll
  for (int pf = 0; pf < 4; ++pf)
#pragma unroll
    for (int r = 0; r < 8; ++r)
      pw[(r + 8 * half) * 72 + pf * 16 + l16] = (__bf16)(e[pf][r] / rsum[r]);
  // wave-private region: same-wave LDS ops are in-order; no block barrier needed

  // A-fragments of probs (K = P = 64)
  v16bf ap[2];
#pragma unroll
  for (int kb = 0; kb < 2; ++kb)
    ap[kb] = frag_a_lds(pw + l16 * 72 + kb * 32, half);

  // out[n, d] = probs @ vp^T : B[k=p][col=d] = svp[d][p]
  v8f ov[4];
#pragma unroll
  for (int df = 0; df < 4; ++df)
    ov[df] = (v8f){0.f, 0.f, 0.f, 0.f, 0.f, 0.f, 0.f, 0.f};
#pragma unroll
  for (int df = 0; df < 4; ++df)
#pragma unroll
    for (int kb = 0; kb < 2; ++kb) {
      v16bf bk = frag_b_lds(svp + (df * 16 + l16) * 72 + kb * 32, half);
      ov[df] = __builtin_amdgcn_wmma_f32_16x16x32_bf16(
          false, ap[kb], false, bk, (short)0, ov[df], false, false);
    }

  // scrambled permute epilogue: i = d*H*N + h*N + n; row=i/C, col=i%C
#pragma unroll
  for (int df = 0; df < 4; ++df)
#pragma unroll
    for (int r = 0; r < 8; ++r) {
      const int n = blockIdx.x * 128 + wid * 16 + r + 8 * half;
      const int d = df * 16 + l16;
      const int i = d * HNT + h * NN + n;
      const int row = i >> 9, col = i & 511;
      x_sa[((long long)b * NN + row) * CC + col] = (__bf16)ov[df][r];
    }
}

// ---------------------------------------------------------------------------
// channel-attention row softmax: attn = softmax(scores * temperature), bf16 out
// one block per row (B*C rows of 512)
// ---------------------------------------------------------------------------
__global__ __launch_bounds__(256) void softmax_ca_kernel(
    const float* __restrict__ sc, const float* __restrict__ temp,
    __bf16* __restrict__ attn) {
  const int row = blockIdx.x;
  const int t = threadIdx.x;
  const float* r = sc + (long long)row * 512;
  const float tv = temp[0];
  const float v0 = r[t] * tv, v1 = r[t + 256] * tv;
  __shared__ float red[256];
  __shared__ float bm, bs;
  red[t] = fmaxf(v0, v1);
  __syncthreads();
  for (int o = 128; o; o >>= 1) {
    if (t < o) red[t] = fmaxf(red[t], red[t + o]);
    __syncthreads();
  }
  if (t == 0) bm = red[0];
  __syncthreads();
  const float e0 = __expf(v0 - bm), e1 = __expf(v1 - bm);
  red[t] = e0 + e1;
  __syncthreads();
  for (int o = 128; o; o >>= 1) {
    if (t < o) red[t] += red[t + o];
    __syncthreads();
  }
  if (t == 0) bs = 1.f / red[0];
  __syncthreads();
  attn[(long long)row * 512 + t]       = (__bf16)(e0 * bs);
  attn[(long long)row * 512 + t + 256] = (__bf16)(e1 * bs);
}

// ---------------------------------------------------------------------------
// launcher
// ---------------------------------------------------------------------------
extern "C" void kernel_launch(void* const* d_in, const int* in_sizes, int n_in,
                              void* d_out, int out_size, void* d_ws, size_t ws_size,
                              hipStream_t stream) {
  (void)in_sizes; (void)n_in; (void)out_size; (void)ws_size;
  const float* x     = (const float*)d_in[0];  // [8,4096,512]
  const float* w_qkv = (const float*)d_in[1];  // [1536,512]
  const float* w_e   = (const float*)d_in[2];  // [64,4096]
  const float* b_e   = (const float*)d_in[3];  // [64]
  const float* temp  = (const float*)d_in[4];  // [1]
  const float* temp2 = (const float*)d_in[5];  // [8]
  const float* w_o1  = (const float*)d_in[6];  // [256,512]
  const float* b_o1  = (const float*)d_in[7];  // [256]
  const float* w_o2  = (const float*)d_in[8];  // [256,512]
  const float* b_o2  = (const float*)d_in[9];  // [256]
  float* out = (float*)d_out;                  // [8,4096,512]

  // ---- workspace carve-up (256B aligned) ----
  char* w = (char*)d_ws;
  size_t off = 0;
  auto take = [&](size_t bytes) {
    size_t r = off;
    off += (bytes + 255) & ~(size_t)255;
    return r;
  };
  __bf16* x_bf    = (__bf16*)(w + take(BNT * CC * 2));            // 32 MB
  __bf16* wqkv_bf = (__bf16*)(w + take(3LL * CC * CC * 2));       // 1.5 MB
  __bf16* we_bf   = (__bf16*)(w + take((long long)PP * NN * 2));  // 512 KB
  __bf16* wo1_bf  = (__bf16*)(w + take((long long)(CC / 2) * CC * 2));
  __bf16* wo2_bf  = (__bf16*)(w + take((long long)(CC / 2) * CC * 2));
  __bf16* qkv_bf  = (__bf16*)(w + take(BNT * 3 * CC * 2));        // 96 MB
  float*  invn    = (float*) (w + take(4096 * 4));
  __bf16* qn_bf   = (__bf16*)(w + take(BNT * CC * 2));            // [B,H,N,D]
  __bf16* kT_bf   = (__bf16*)(w + take(BNT * CC * 2));            // [B,H,D,N]
  __bf16* vT_bf   = (__bf16*)(w + take(BNT * CC * 2));            // [B,H,D,N]
  __bf16* xT_bf   = (__bf16*)(w + take(BNT * CC * 2));            // [B,C,N]
  __bf16* kpT_bf  = (__bf16*)(w + take(64LL * 64 * 64 * 2));      // [B,H,P,D]
  __bf16* vp_bf   = (__bf16*)(w + take(64LL * 64 * 64 * 2));      // [B,H,D,P]
  float*  scores  = (float*) (w + take(8LL * 512 * 512 * 4));     // [B,C,C]
  __bf16* attn_bf = (__bf16*)(w + take(8LL * 512 * 512 * 2));     // [B,C,C]
  // qkv_bf is dead after repack; reuse it for x_sa / x_ca (each 32 MB < 96 MB)
  __bf16* x_sa_bf = qkv_bf;
  __bf16* x_ca_bf = qkv_bf + BNT * CC;

  // 1) converts
  cvt_f32_bf16_kernel<<<4096, 256, 0, stream>>>(x, x_bf, BNT * CC);
  cvt_f32_bf16_kernel<<<1024, 256, 0, stream>>>(w_qkv, wqkv_bf, 3LL * CC * CC);
  cvt_f32_bf16_kernel<<<512,  256, 0, stream>>>(w_e, we_bf, (long long)PP * NN);
  cvt_f32_bf16_kernel<<<256,  256, 0, stream>>>(w_o1, wo1_bf, (long long)(CC / 2) * CC);
  cvt_f32_bf16_kernel<<<256,  256, 0, stream>>>(w_o2, wo2_bf, (long long)(CC / 2) * CC);

  // 2) qkv = x @ w_qkv^T   [32768 x 1536 x 512]
  gemm_bt_kernel<0><<<dim3(12, 256, 1), 256, 0, stream>>>(
      x_bf, wqkv_bf, qkv_bf, nullptr,
      (int)BNT, 1536, 512, 512, 512, 1536, 0, 0, 0);

  // 3) q norms + repack
  qnorm_kernel<<<4096, 256, 0, stream>>>(qkv_bf, invn);
  repack_kernel<<<(int)((BNT * CC + 255) / 256), 256, 0, stream>>>(
      qkv_bf, x_bf, invn, qn_bf, kT_bf, vT_bf, xT_bf);

  // 4) kp^T = (k_t @ w_e^T)^T (+b_e) and vp = v_t @ w_e^T (+b_e)
  //    64 batches of [64 x 64 x 4096]
  gemm_bt_kernel<1><<<dim3(1, 1, 64), 256, 0, stream>>>(
      kT_bf, we_bf, kpT_bf, b_e,
      64, 64, NN, NN, NN, 64, 64LL * NN, 0, 64 * 64);
  gemm_bt_kernel<0><<<dim3(1, 1, 64), 256, 0, stream>>>(
      vT_bf, we_bf, vp_bf, b_e,
      64, 64, NN, NN, NN, 64, 64LL * NN, 0, 64 * 64);

  // 5) fused projected spatial attention + scrambled permute
  attn_sa_kernel<<<dim3(NN / 128, BB * HH), 256, 0, stream>>>(
      qn_bf, kpT_bf, vp_bf, temp2, x_sa_bf);

  // 6) channel attention: scores = x^T x  [8 x (512 x 512 x 4096)], fp32 out
  gemm_bt_kernel<2><<<dim3(4, 4, 8), 256, 0, stream>>>(
      xT_bf, xT_bf, scores, nullptr,
      512, 512, NN, NN, NN, 512, (long long)CC * NN, (long long)CC * NN, 512LL * 512);
  softmax_ca_kernel<<<BB * CC, 256, 0, stream>>>(scores, temp, attn_bf);
  //    x_ca = x @ attn^T  [8 x (4096 x 512 x 512)]
  gemm_bt_kernel<0><<<dim3(4, 32, 8), 256, 0, stream>>>(
      x_bf, attn_bf, x_ca_bf, nullptr,
      NN, 512, 512, 512, 512, 512, (long long)NN * CC, 512LL * 512, (long long)NN * CC);

  // 7) out[:, :, :256]  = x_sa @ w_o1^T + b_o1
  //    out[:, :, 256:]  = x_ca @ w_o2^T + b_o2   (ldc = 512, fp32 into d_out)
  gemm_bt_kernel<2><<<dim3(2, 256, 1), 256, 0, stream>>>(
      x_sa_bf, wo1_bf, out, b_o1,
      (int)BNT, 256, 512, 512, 512, 512, 0, 0, 0);
  gemm_bt_kernel<2><<<dim3(2, 256, 1), 256, 0, stream>>>(
      x_ca_bf, wo2_bf, out + 256, b_o2,
      (int)BNT, 256, 512, 512, 512, 512, 0, 0, 0);
}
